// PDECausalModel_9045201125625
// MI455X (gfx1250) — matrix operand order repaired and checked by reading, not compiled, not run on hardware
//
#include <hip/hip_runtime.h>
#include <hip/hip_bf16.h>
#include <cstdint>

// ---------------------------------------------------------------------------
// Model dims (fixed by the reference)
// ---------------------------------------------------------------------------
#define S_TOK 4096   // T*P = 16*256
#define P_BLK 256
#define DMODEL 512
#define NHEAD 8
#define HDIM 64
#define FFN 2048
#define NLAYER 4
#define CIO 6

typedef _Float16 half_t;
typedef __attribute__((ext_vector_type(16))) half_t v16h;
typedef __attribute__((ext_vector_type(8)))  half_t v8h;
typedef __attribute__((ext_vector_type(8)))  float  v8f;

// ---------------------------------------------------------------------------
// WMMA helpers (CDNA5 wave32, V_WMMA_F32_16X16X32_F16)
// ---------------------------------------------------------------------------
__device__ __forceinline__ v8f zero8() {
    v8f z = {0.f, 0.f, 0.f, 0.f, 0.f, 0.f, 0.f, 0.f};
    return z;
}

__device__ __forceinline__ v8f wmma32(v16h a, v16h b, v8f c) {
    return __builtin_amdgcn_wmma_f32_16x16x32_f16(
        /*neg_a=*/false, a, /*neg_b=*/false, b,
        /*c_mod=*/(short)0, c, /*reuse_a=*/false, /*reuse_b=*/false);
}

__device__ __forceinline__ v16h cat16(v8h lo, v8h hi) {
    v16h r;
#pragma unroll
    for (int i = 0; i < 8; ++i) { r[i] = lo[i]; r[i + 8] = hi[i]; }
    return r;
}

// A-matrix 16x32 (MxK), row-major source with leading dim ld.
// ISA layout: lane l holds row m=l&15; halves 0..7 -> K = hb+0..7,
// halves 8..15 -> K = 16+hb+0..7, where hb = (l>>4)*8.
__device__ __forceinline__ v16h load_A16x32(const half_t* base, int ld) {
    int lane = threadIdx.x & 31;
    int m  = lane & 15;
    int hb = (lane >> 4) << 3;
    const half_t* p = base + m * ld + hb;
    v8h lo = *(const v8h*)(p);
    v8h hi = *(const v8h*)(p + 16);
    return cat16(lo, hi);
}

// B-matrix 32x16 (KxN) sourced from an N-major array (src[n*ld + k]).
// ISA layout: lane l holds col n=l&15; halves 0..7 -> K = kb+0..7,
// halves 8..15 -> K = 16+kb+0..7, where kb = (l>>4)*8.
__device__ __forceinline__ v16h load_Bn16x32(const half_t* base, int ld) {
    int lane = threadIdx.x & 31;
    int n  = lane & 15;
    int kb = (lane >> 4) << 3;
    const half_t* p = base + n * ld + kb;
    v8h lo = *(const v8h*)(p);
    v8h hi = *(const v8h*)(p + 16);
    return cat16(lo, hi);
}

// ---------------------------------------------------------------------------
// Weight convert: f32 (K x N, row-major) -> f16 transposed (N x K, row-major)
// ---------------------------------------------------------------------------
__global__ void wconv_t_kernel(const float* __restrict__ w, half_t* __restrict__ wt,
                               int K, int N) {
    int i  = blockIdx.x * 256 + threadIdx.x;   // grid sized exactly K*N/256
    int kk = i / N;
    int n  = i - kk * N;
    wt[(size_t)n * K + kk] = (half_t)w[i];
}

// ---------------------------------------------------------------------------
// Encoder: h[s,d] = sum_c x[s,c] * W_enc[c,d] + b_enc[d]    (C=6, trivial)
// ---------------------------------------------------------------------------
__global__ void encode_kernel(const float* __restrict__ x, const float* __restrict__ We,
                              const float* __restrict__ be, float* __restrict__ h) {
    int i = blockIdx.x * 256 + threadIdx.x;    // S*D threads
    int s = i >> 9;
    int d = i & (DMODEL - 1);
    const float* xp = x + (size_t)s * CIO;
    float acc = be[d];
#pragma unroll
    for (int c = 0; c < CIO; ++c) acc += xp[c] * We[c * DMODEL + d];
    h[i] = acc;
}

// ---------------------------------------------------------------------------
// RMSNorm: a = h * rsqrt(mean(h^2)+eps) * w  -> f16
// one block per row (256 threads, D=512 -> 2 elems/thread)
// ---------------------------------------------------------------------------
__global__ void rmsnorm_kernel(const float* __restrict__ h, const float* __restrict__ w,
                               half_t* __restrict__ out) {
    __shared__ float red[256];
    int s = blockIdx.x, t = threadIdx.x;
    const float* hp = h + (size_t)s * DMODEL;
    float x0 = hp[t], x1 = hp[t + 256];
    red[t] = x0 * x0 + x1 * x1;
    __syncthreads();
    for (int off = 128; off > 0; off >>= 1) {
        if (t < off) red[t] += red[t + off];
        __syncthreads();
    }
    float inv = rsqrtf(red[0] * (1.0f / DMODEL) + 1e-5f);
    out[(size_t)s * DMODEL + t]       = (half_t)(x0 * inv * w[t]);
    out[(size_t)s * DMODEL + t + 256] = (half_t)(x1 * inv * w[t + 256]);
}

// ---------------------------------------------------------------------------
// GEMM: Y[s,n] = sum_k Act[s,k] * Wt[n,k]
//   Act: (S x K) f16 row-major; Wt: (N x K) f16 row-major (pre-transposed)
//   Each wave computes a 32x64 output block (2x4 tiles of 16x16).
//   A fragments live across the K-step (reused 4x); B fragments are
//   transient (reused 2x) -> ~110 live VGPRs, no spills.
//   mode 0: store f16 row-major -> yh
//   mode 1: accumulate f32 -> hres (residual add)
//   mode 2: store f16 TRANSPOSED -> yh[n*S_TOK + m]   (for V^T)
// ---------------------------------------------------------------------------
__global__ void gemm_wmma_kernel(const half_t* __restrict__ act,
                                 const half_t* __restrict__ wt,
                                 half_t* __restrict__ yh,
                                 float* __restrict__ hres,
                                 int K, int N, int mode) {
    int wave  = threadIdx.x >> 5;
    int lane  = threadIdx.x & 31;
    int task  = blockIdx.x * 8 + wave;
    int nblk  = N >> 6;
    int row_b = task / nblk;
    int colg  = task - row_b * nblk;

    const half_t* abase = act + (size_t)row_b * 32 * K;
    const half_t* b0    = wt + (size_t)colg * 64 * K;

    v8f acc[8];
#pragma unroll
    for (int i = 0; i < 8; ++i) acc[i] = zero8();

    for (int kk = 0; kk < K; kk += 32) {
        v16h a0 = load_A16x32(abase + kk, K);
        v16h a1 = load_A16x32(abase + (size_t)16 * K + kk, K);
        // speculative prefetch of the A stream eight K-blocks (256 halves) ahead (near scope)
        __builtin_prefetch(abase + (size_t)(lane & 15) * K + kk + 256, 0, 3);
#pragma unroll
        for (int t = 0; t < 4; ++t) {
            v16h b = load_Bn16x32(b0 + (size_t)(t * 16) * K + kk, K);
            acc[t]     = wmma32(a0, b, acc[t]);
            acc[4 + t] = wmma32(a1, b, acc[4 + t]);
        }
    }

    int nn = lane & 15;
    int mh = (lane >> 4) << 3;
    if (mode == 0) {
#pragma unroll
        for (int mi = 0; mi < 2; ++mi)
#pragma unroll
            for (int t = 0; t < 4; ++t)
#pragma unroll
                for (int r = 0; r < 8; ++r) {
                    int m = row_b * 32 + mi * 16 + mh + r;
                    int n = colg * 64 + t * 16 + nn;
                    yh[(size_t)m * N + n] = (half_t)acc[mi * 4 + t][r];
                }
    } else if (mode == 1) {
#pragma unroll
        for (int mi = 0; mi < 2; ++mi)
#pragma unroll
            for (int t = 0; t < 4; ++t)
#pragma unroll
                for (int r = 0; r < 8; ++r) {
                    int m = row_b * 32 + mi * 16 + mh + r;
                    int n = colg * 64 + t * 16 + nn;
                    hres[(size_t)m * N + n] += acc[mi * 4 + t][r];
                }
    } else {  // transposed f16 store: yh[n * S_TOK + m]
#pragma unroll
        for (int mi = 0; mi < 2; ++mi)
#pragma unroll
            for (int t = 0; t < 4; ++t)
#pragma unroll
                for (int r = 0; r < 8; ++r) {
                    int m = row_b * 32 + mi * 16 + mh + r;
                    int n = colg * 64 + t * 16 + nn;
                    yh[(size_t)n * S_TOK + m] = (half_t)acc[mi * 4 + t][r];
                }
    }
}

// ---------------------------------------------------------------------------
// RoPE (in place on q,k f16): pairs (i, i+32) within each head, angle s*theta^(-i/32)
// ---------------------------------------------------------------------------
__global__ void rope_kernel(half_t* __restrict__ q, half_t* __restrict__ k) {
    int idx = blockIdx.x * 256 + threadIdx.x;   // S*NH*32 threads
    int i    = idx & 31;
    int head = (idx >> 5) & (NHEAD - 1);
    int s    = idx >> 8;
    float inv = __powf(10000.0f, -(float)i * (1.0f / 32.0f));
    float ang = (float)s * inv;
    float sn, cs;
    __sincosf(ang, &sn, &cs);
    size_t base = (size_t)s * DMODEL + head * HDIM;
    float q0 = (float)q[base + i], q1 = (float)q[base + i + 32];
    q[base + i]      = (half_t)(q0 * cs - q1 * sn);
    q[base + i + 32] = (half_t)(q1 * cs + q0 * sn);
    float k0 = (float)k[base + i], k1 = (float)k[base + i + 32];
    k[base + i]      = (half_t)(k0 * cs - k1 * sn);
    k[base + i + 32] = (half_t)(k1 * cs + k0 * sn);
}

// ---------------------------------------------------------------------------
// Flash attention, block-causal (mask granularity P=256 => no partial masks).
// One wave per (16-row tile, head). 8 waves/block. f32 online softmax.
// V is consumed TRANSPOSED (vT: D x S, f16) so all WMMA operands use the
// contiguous two-b128 N-major loader.
// ---------------------------------------------------------------------------
__global__ void attn_kernel(const half_t* __restrict__ q, const half_t* __restrict__ k,
                            const half_t* __restrict__ vT, half_t* __restrict__ o) {
    __shared__ __attribute__((aligned(16))) half_t pbuf[8 * 16 * 32];
    int wave = threadIdx.x >> 5;
    int lane = threadIdx.x & 31;
    int task = blockIdx.x * 8 + wave;
    int head = task & (NHEAD - 1);
    int rt   = task >> 3;
    int srow = rt * 16;
    int ncols = ((srow >> 8) + 1) << 8;          // (block+1)*256 visible keys

    const int ld = DMODEL;                        // stride between tokens (q,k)
    const half_t* qb  = q + (size_t)srow * ld + head * HDIM;
    const half_t* kb  = k + head * HDIM;
    const half_t* vtb = vT + (size_t)head * HDIM * S_TOK;  // rows d, cols s

    v16h qa0 = load_A16x32(qb, ld);               // K-dim 0..31
    v16h qa1 = load_A16x32(qb + 32, ld);          // K-dim 32..63

    float m_run[8], l_run[8];
#pragma unroll
    for (int r = 0; r < 8; ++r) { m_run[r] = -1e30f; l_run[r] = 0.f; }
    v8f acc0 = zero8(), acc1 = zero8(), acc2 = zero8(), acc3 = zero8();

    const float scale = 0.125f;                   // 1/sqrt(64)
    half_t* pb = pbuf + wave * (16 * 32);
    int nn = lane & 15;
    int mb = (lane >> 4) << 3;

    for (int j0 = 0; j0 < ncols; j0 += 32) {
        // ---- scores: two 16x16 tiles (cols j0..j0+15, j0+16..j0+31)
        const half_t* k0 = kb + (size_t)j0 * ld;
        const half_t* k1 = kb + (size_t)(j0 + 16) * ld;
        v8f s0 = zero8(), s1 = zero8();
        s0 = wmma32(qa0, load_Bn16x32(k0, ld), s0);
        s0 = wmma32(qa1, load_Bn16x32(k0 + 32, ld), s0);
        s1 = wmma32(qa0, load_Bn16x32(k1, ld), s1);
        s1 = wmma32(qa1, load_Bn16x32(k1 + 32, ld), s1);

        // ---- online softmax over the 32 new columns
        float p0[8], p1[8];
#pragma unroll
        for (int r = 0; r < 8; ++r) {
            float t0 = s0[r] * scale, t1 = s1[r] * scale;
            float mx = fmaxf(t0, t1);
            mx = fmaxf(mx, __shfl_xor(mx, 1));
            mx = fmaxf(mx, __shfl_xor(mx, 2));
            mx = fmaxf(mx, __shfl_xor(mx, 4));
            mx = fmaxf(mx, __shfl_xor(mx, 8));
            float mnew = fmaxf(m_run[r], mx);
            float corr = __expf(m_run[r] - mnew);
            float e0 = __expf(t0 - mnew);
            float e1 = __expf(t1 - mnew);
            float rs = e0 + e1;
            rs += __shfl_xor(rs, 1);
            rs += __shfl_xor(rs, 2);
            rs += __shfl_xor(rs, 4);
            rs += __shfl_xor(rs, 8);
            l_run[r] = l_run[r] * corr + rs;
            m_run[r] = mnew;
            acc0[r] *= corr; acc1[r] *= corr; acc2[r] *= corr; acc3[r] *= corr;
            p0[r] = e0; p1[r] = e1;
        }

        // ---- C-layout P -> row-major LDS tile -> A-layout reload (same wave,
        //      LDS ops are in-order within a wave)
#pragma unroll
        for (int r = 0; r < 8; ++r) {
            pb[(mb + r) * 32 + nn]      = (half_t)p0[r];
            pb[(mb + r) * 32 + 16 + nn] = (half_t)p1[r];
        }
        v16h pa = load_A16x32(pb, 32);

        // ---- P(16x32) @ V(32x64): 4 WMMAs, B from vT (contiguous loads)
        acc0 = wmma32(pa, load_Bn16x32(vtb + (size_t)0  * S_TOK + j0, S_TOK), acc0);
        acc1 = wmma32(pa, load_Bn16x32(vtb + (size_t)16 * S_TOK + j0, S_TOK), acc1);
        acc2 = wmma32(pa, load_Bn16x32(vtb + (size_t)32 * S_TOK + j0, S_TOK), acc2);
        acc3 = wmma32(pa, load_Bn16x32(vtb + (size_t)48 * S_TOK + j0, S_TOK), acc3);
    }

    // ---- normalize + store o[s, head, d] f16
#pragma unroll
    for (int r = 0; r < 8; ++r) {
        float linv = 1.0f / l_run[r];
        size_t row = (size_t)(srow + mb + r) * DMODEL + head * HDIM + nn;
        o[row]      = (half_t)(acc0[r] * linv);
        o[row + 16] = (half_t)(acc1[r] * linv);
        o[row + 32] = (half_t)(acc2[r] * linv);
        o[row + 48] = (half_t)(acc3[r] * linv);
    }
}

// ---------------------------------------------------------------------------
// SwiGLU: g = silu(g) * u  (in place on g)
// ---------------------------------------------------------------------------
__global__ void swiglu_kernel(half_t* __restrict__ g, const half_t* __restrict__ u) {
    size_t i = (size_t)blockIdx.x * 256 + threadIdx.x;  // S*F threads exact
    float gv = (float)g[i];
    float uv = (float)u[i];
    float sv = gv / (1.0f + __expf(-gv));
    g[i] = (half_t)(sv * uv);
}

// ---------------------------------------------------------------------------
// Decoder: out[s,c] = sum_d a[s,d] * W_dec[d,c] + b_dec[c]
// ---------------------------------------------------------------------------
__global__ void decode_kernel(const half_t* __restrict__ a, const float* __restrict__ Wd,
                              const float* __restrict__ bd, float* __restrict__ out) {
    int idx = blockIdx.x * 256 + threadIdx.x;  // S*C = 24576 threads exact
    int s = idx / CIO;
    int c = idx - s * CIO;
    const half_t* ap = a + (size_t)s * DMODEL;
    float acc = bd[c];
#pragma unroll 8
    for (int d = 0; d < DMODEL; ++d) acc += (float)ap[d] * Wd[d * CIO + c];
    out[idx] = acc;
}

// ---------------------------------------------------------------------------
// Host orchestration
// ---------------------------------------------------------------------------
extern "C" void kernel_launch(void* const* d_in, const int* in_sizes, int n_in,
                              void* d_out, int out_size, void* d_ws, size_t ws_size,
                              hipStream_t stream) {
    (void)in_sizes; (void)n_in; (void)out_size; (void)ws_size;
    const float* x     = (const float*)d_in[0];
    const float* W_enc = (const float*)d_in[1];
    const float* b_enc = (const float*)d_in[2];
    const float* Wq    = (const float*)d_in[3];
    const float* Wk    = (const float*)d_in[4];
    const float* Wv    = (const float*)d_in[5];
    const float* Wo    = (const float*)d_in[6];
    const float* Wg    = (const float*)d_in[7];
    const float* Wu    = (const float*)d_in[8];
    const float* Wd    = (const float*)d_in[9];
    const float* ln1   = (const float*)d_in[10];
    const float* ln2   = (const float*)d_in[11];
    const float* ln_f  = (const float*)d_in[12];
    const float* W_dec = (const float*)d_in[13];
    const float* b_dec = (const float*)d_in[14];
    float* out = (float*)d_out;

    // ---- workspace carve (all aligned 256B)
    char* ws = (char*)d_ws;
    size_t off = 0;
    auto carve = [&](size_t bytes) -> char* {
        char* p = ws + off;
        off += (bytes + 255) & ~(size_t)255;
        return p;
    };
    const size_t DD = (size_t)DMODEL * DMODEL;   // 262144
    const size_t DF = (size_t)DMODEL * FFN;      // 1048576
    const size_t SD = (size_t)S_TOK * DMODEL;    // 2097152
    const size_t SF = (size_t)S_TOK * FFN;       // 8388608

    half_t* wq_t = (half_t*)carve(NLAYER * DD * sizeof(half_t));
    half_t* wk_t = (half_t*)carve(NLAYER * DD * sizeof(half_t));
    half_t* wv_t = (half_t*)carve(NLAYER * DD * sizeof(half_t));
    half_t* wo_t = (half_t*)carve(NLAYER * DD * sizeof(half_t));
    half_t* wg_t = (half_t*)carve(NLAYER * DF * sizeof(half_t));
    half_t* wu_t = (half_t*)carve(NLAYER * DF * sizeof(half_t));
    half_t* wd_t = (half_t*)carve(NLAYER * DF * sizeof(half_t));
    float*  h    = (float*)carve(SD * sizeof(float));
    half_t* aB   = (half_t*)carve(SD * sizeof(half_t));
    half_t* qB   = (half_t*)carve(SD * sizeof(half_t));
    half_t* kB   = (half_t*)carve(SD * sizeof(half_t));
    half_t* vTB  = (half_t*)carve(SD * sizeof(half_t));   // V transposed: D x S
    half_t* oB   = (half_t*)carve(SD * sizeof(half_t));
    half_t* gB   = (half_t*)carve(SF * sizeof(half_t));
    half_t* uB   = (half_t*)carve(SF * sizeof(half_t));

    // ---- f32 -> f16 transposed weight conversion (deterministic, every call)
    for (int l = 0; l < NLAYER; ++l) {
        wconv_t_kernel<<<DD / 256, 256, 0, stream>>>(Wq + l * DD, wq_t + l * DD, DMODEL, DMODEL);
        wconv_t_kernel<<<DD / 256, 256, 0, stream>>>(Wk + l * DD, wk_t + l * DD, DMODEL, DMODEL);
        wconv_t_kernel<<<DD / 256, 256, 0, stream>>>(Wv + l * DD, wv_t + l * DD, DMODEL, DMODEL);
        wconv_t_kernel<<<DD / 256, 256, 0, stream>>>(Wo + l * DD, wo_t + l * DD, DMODEL, DMODEL);
        wconv_t_kernel<<<DF / 256, 256, 0, stream>>>(Wg + l * DF, wg_t + l * DF, DMODEL, FFN);
        wconv_t_kernel<<<DF / 256, 256, 0, stream>>>(Wu + l * DF, wu_t + l * DF, DMODEL, FFN);
        wconv_t_kernel<<<DF / 256, 256, 0, stream>>>(Wd + l * DF, wd_t + l * DF, FFN, DMODEL);
    }

    // ---- encoder
    encode_kernel<<<SD / 256, 256, 0, stream>>>(x, W_enc, b_enc, h);

    // ---- transformer layers
    // 32x64 wave tiles: tasks = (S/32)*(N/64), 8 waves/block (exact division)
    const int gemmD_blocks = (S_TOK / 32) * (DMODEL / 64) / 8;  // 128
    const int gemmF_blocks = (S_TOK / 32) * (FFN / 64) / 8;     // 512
    for (int l = 0; l < NLAYER; ++l) {
        rmsnorm_kernel<<<S_TOK, 256, 0, stream>>>(h, ln1 + l * DMODEL, aB);
        gemm_wmma_kernel<<<gemmD_blocks, 256, 0, stream>>>(aB, wq_t + l * DD, qB, nullptr, DMODEL, DMODEL, 0);
        gemm_wmma_kernel<<<gemmD_blocks, 256, 0, stream>>>(aB, wk_t + l * DD, kB, nullptr, DMODEL, DMODEL, 0);
        gemm_wmma_kernel<<<gemmD_blocks, 256, 0, stream>>>(aB, wv_t + l * DD, vTB, nullptr, DMODEL, DMODEL, 2);
        rope_kernel<<<(S_TOK * NHEAD * 32) / 256, 256, 0, stream>>>(qB, kB);
        attn_kernel<<<(S_TOK / 16) * NHEAD / 8, 256, 0, stream>>>(qB, kB, vTB, oB);
        gemm_wmma_kernel<<<gemmD_blocks, 256, 0, stream>>>(oB, wo_t + l * DD, nullptr, h, DMODEL, DMODEL, 1);
        rmsnorm_kernel<<<S_TOK, 256, 0, stream>>>(h, ln2 + l * DMODEL, aB);
        gemm_wmma_kernel<<<gemmF_blocks, 256, 0, stream>>>(aB, wg_t + l * DF, gB, nullptr, DMODEL, FFN, 0);
        gemm_wmma_kernel<<<gemmF_blocks, 256, 0, stream>>>(aB, wu_t + l * DF, uB, nullptr, DMODEL, FFN, 0);
        swiglu_kernel<<<SF / 256, 256, 0, stream>>>(gB, uB);
        gemm_wmma_kernel<<<gemmD_blocks, 256, 0, stream>>>(gB, wd_t + l * DF, nullptr, h, FFN, DMODEL, 1);
    }

    // ---- final norm + decoder
    rmsnorm_kernel<<<S_TOK, 256, 0, stream>>>(h, ln_f, aB);
    decode_kernel<<<(S_TOK * CIO) / 256, 256, 0, stream>>>(aB, W_dec, b_dec, out);
}
